// PointNet2_34574486733459
// MI455X (gfx1250) — matrix operand order repaired and checked by reading, not compiled, not run on hardware
//
#include <hip/hip_runtime.h>

typedef __attribute__((ext_vector_type(16))) __bf16 v16bf;
typedef __attribute__((ext_vector_type(8)))  float  v8f;
typedef unsigned short u16;
typedef unsigned int   u32;

#define PN_B  16
#define PN_N  16384
#define PN_NS 16

__device__ __forceinline__ u16 f2bf(float f) {
  u32 u = __float_as_uint(f);
  u32 r = u + 0x7FFFu + ((u >> 16) & 1u);   // round-to-nearest-even
  return (u16)(r >> 16);
}
__device__ __forceinline__ float bf2f(u16 h) { return __uint_as_float(((u32)h) << 16); }

// ---------------------------------------------------------------- utility kernels

__global__ void k_transpose_l0(const float* __restrict__ pts, float* __restrict__ out, int total) {
  int t = blockIdx.x * blockDim.x + threadIdx.x;
  if (t >= total) return;                       // out (B,3,N) = transpose of pts (B,N,3)
  int b = t / (3 * PN_N); int rm = t % (3 * PN_N);
  int c = rm / PN_N;      int j  = rm % PN_N;
  out[t] = pts[((size_t)b * PN_N + j) * 3 + c];
}

__global__ void k_convert_w(const float* __restrict__ W, u16* __restrict__ out,
                            int cout, int cin, int Kpad, int total) {
  int t = blockIdx.x * blockDim.x + threadIdx.x;
  if (t >= total) return;
  int row = t / Kpad, col = t % Kpad;
  out[t] = (row < cout && col < cin) ? f2bf(W[(size_t)row * cin + col]) : (u16)0;
}

// pack offset(3x128) + cls(1x128) heads into one padded 64x128 bf16 weight + f32 bias
__global__ void k_pack_heads(const float* __restrict__ Wo, const float* __restrict__ bo,
                             const float* __restrict__ Wc, const float* __restrict__ bc,
                             u16* __restrict__ Wh, float* __restrict__ bh) {
  int t = blockIdx.x * blockDim.x + threadIdx.x;
  if (t < 64 * 128) {
    int row = t / 128, col = t % 128;
    float v = 0.0f;
    if (row < 3) v = Wo[row * 128 + col];
    else if (row == 3) v = Wc[col];
    Wh[t] = f2bf(v);
  }
  if (t < 64) {
    float v = 0.0f;
    if (t < 3) v = bo[t]; else if (t == 3) v = bc[0];
    bh[t] = v;
  }
}

// ---------------------------------------------------------------- furthest point sampling
__global__ void k_fps(const float* __restrict__ xyz, int* __restrict__ idx, int N, int np) {
  extern __shared__ char fps_smem[];
  float* mind = (float*)fps_smem;
  float* rv   = mind + N;
  int*   ri   = (int*)(rv + blockDim.x);
  __shared__ int s_last;
  const int b = blockIdx.x, t = threadIdx.x, bd = blockDim.x;
  const float* px = xyz + (size_t)b * N * 3;
  for (int j = t; j < N; j += bd) mind[j] = 1e10f;
  if (t == 0) s_last = 0;
  __syncthreads();
  for (int it = 0; it < np; ++it) {
    int last = s_last;
    if (t == 0) idx[b * np + it] = last;
    float cx = px[last * 3], cy = px[last * 3 + 1], cz = px[last * 3 + 2];
    float bestv = -1.0f; int besti = 0;
    for (int j = t; j < N; j += bd) {
      float dx = px[j * 3] - cx, dy = px[j * 3 + 1] - cy, dz = px[j * 3 + 2] - cz;
      float d = dx * dx + dy * dy + dz * dz;
      float m = mind[j]; m = d < m ? d : m; mind[j] = m;
      if (m > bestv) { bestv = m; besti = j; }
    }
    rv[t] = bestv; ri[t] = besti;
    __syncthreads();
    for (int s = bd >> 1; s > 0; s >>= 1) {
      if (t < s) {
        float v2 = rv[t + s]; int i2 = ri[t + s];
        if (v2 > rv[t] || (v2 == rv[t] && i2 < ri[t])) { rv[t] = v2; ri[t] = i2; }
      }
      __syncthreads();
    }
    if (t == 0) s_last = ri[0];
    __syncthreads();
  }
}

__global__ void k_gather_xyz(const float* __restrict__ xyz, const int* __restrict__ idx,
                             float* __restrict__ nxyz, int S, int Nl, int total) {
  int t = blockIdx.x * blockDim.x + threadIdx.x;
  if (t >= total) return;
  int b = t / S;
  int gi = idx[t];
  const float* p = xyz + ((size_t)b * Nl + gi) * 3;
  nxyz[t * 3 + 0] = p[0]; nxyz[t * 3 + 1] = p[1]; nxyz[t * 3 + 2] = p[2];
}

__global__ void k_ballquery(const float* __restrict__ xyz, const float* __restrict__ nxyz,
                            int* __restrict__ gidx, int Nl, int S, float r2, int total) {
  int t = blockIdx.x * blockDim.x + threadIdx.x;
  if (t >= total) return;
  int b = t / S;
  float cx = nxyz[t * 3], cy = nxyz[t * 3 + 1], cz = nxyz[t * 3 + 2];
  const float* px = xyz + (size_t)b * Nl * 3;
  int cnt = 0, first = 0; bool hasfirst = false;
  for (int nn = 0; nn < Nl; ++nn) {
    float dx = px[nn * 3] - cx, dy = px[nn * 3 + 1] - cy, dz = px[nn * 3 + 2] - cz;
    float d = dx * dx + dy * dy + dz * dz;
    if (d < r2) {
      if (!hasfirst) { first = nn; hasfirst = true; }
      if (cnt < PN_NS) gidx[(size_t)t * PN_NS + cnt] = nn;
      ++cnt;
    }
  }
  for (int j = cnt; j < PN_NS; ++j) gidx[(size_t)t * PN_NS + j] = first;
}

__global__ void k_sa_build(const float* __restrict__ xyz, const float* __restrict__ nxyz,
                           const int* __restrict__ gidx, const float* __restrict__ fea,
                           u16* __restrict__ X, int Nl, int S, int C, int Kpad, int ncols) {
  int t = blockIdx.x * blockDim.x + threadIdx.x;
  if (t >= ncols) return;
  int b = t / (S * PN_NS); int rm = t % (S * PN_NS);
  int s = rm / PN_NS;
  int gi = gidx[t];
  const float* pp = xyz + ((size_t)b * Nl + gi) * 3;
  const float* cc = nxyz + ((size_t)b * S + s) * 3;
  X[(size_t)0 * ncols + t] = f2bf(pp[0] - cc[0]);
  X[(size_t)1 * ncols + t] = f2bf(pp[1] - cc[1]);
  X[(size_t)2 * ncols + t] = f2bf(pp[2] - cc[2]);
  for (int c = 0; c < C; ++c)
    X[(size_t)(3 + c) * ncols + t] = f2bf(fea[((size_t)b * C + c) * Nl + gi]);
  for (int k = 3 + C; k < Kpad; ++k) X[(size_t)k * ncols + t] = 0;
}

__global__ void k_fp_build(const float* __restrict__ p1, const float* __restrict__ p2,
                           const float* __restrict__ f1, const float* __restrict__ f2,
                           u16* __restrict__ X, int n, int m, int C1, int C2, int total) {
  int t = blockIdx.x * blockDim.x + threadIdx.x;
  if (t >= total) return;
  int b = t / n, i = t % n;
  int ncols = total;
  float qx = p1[(size_t)t * 3], qy = p1[(size_t)t * 3 + 1], qz = p1[(size_t)t * 3 + 2];
  const float* pm = p2 + (size_t)b * m * 3;
  float d0 = 1e30f, d1 = 1e30f, d2v = 1e30f; int i0 = 0, i1 = 0, i2 = 0;
  for (int j = 0; j < m; ++j) {
    float dx = pm[j * 3] - qx, dy = pm[j * 3 + 1] - qy, dz = pm[j * 3 + 2] - qz;
    float d = dx * dx + dy * dy + dz * dz;
    if (d < d0)       { d2v = d1; i2 = i1; d1 = d0; i1 = i0; d0 = d; i0 = j; }
    else if (d < d1)  { d2v = d1; i2 = i1; d1 = d; i1 = j; }
    else if (d < d2v) { d2v = d; i2 = j; }
  }
  float w0 = 1.0f / (d0 + 1e-8f), w1 = 1.0f / (d1 + 1e-8f), w2 = 1.0f / (d2v + 1e-8f);
  float ws = w0 + w1 + w2; w0 /= ws; w1 /= ws; w2 /= ws;
  const float* fb = f2 + (size_t)b * C2 * m;
  for (int k = 0; k < C2; ++k) {
    const float* fr = fb + (size_t)k * m;
    X[(size_t)k * ncols + t] = f2bf(w0 * fr[i0] + w1 * fr[i1] + w2 * fr[i2]);
  }
  for (int c = 0; c < C1; ++c)
    X[(size_t)(C2 + c) * ncols + t] = f2bf(f1[((size_t)b * C1 + c) * n + i]);
}

// ---------------------------------------------------------------- WMMA GEMM  Y = act(W X + b)
// W: (Mpad x K) bf16 row-major zero-padded; X: (K x ncols) bf16; Y: (M x ncols) bf16.
// Block tile 64(M) x 128(N), BK=32. 8 wave32 waves: wm=wave>>1 row tile, wn=wave&1 col half.
// Each wave: 1 A frag + 4 B frags + 4 WMMAs per K-step.
// A tile staged by async DMA to LDS (ASYNCcnt); B tile transposed in registers to k-pair u32s.
__global__ __launch_bounds__(256)
void k_gemm_bias_relu(const u16* __restrict__ W, const float* __restrict__ bias,
                      const u16* __restrict__ X, u16* __restrict__ Y,
                      int M, int K, int ncols, int relu) {
  __shared__ u32 As[64 * 16];     // 64 rows x 16 k-pairs, row-major
  __shared__ u32 Bs[128 * 16];    // 128 cols x 16 k-pairs, col(n)-major

  const int tid  = threadIdx.x;
  const int lane = tid & 31;
  const int wave = tid >> 5;
  const int wm   = wave >> 1;
  const int wn   = wave & 1;
  const int m0   = blockIdx.y * 64;
  const long long n0 = (long long)blockIdx.x * 128;
  const int hi16 = (lane >> 4) & 1;
  const int l15  = lane & 15;

  // ISA 10.2: flat LDS address low 32 bits == LDS byte offset
  const u32 as_base = (u32)(size_t)(void*)As;

  const u32* W32 = (const u32*)W;
  const int Kh = K >> 1;

  v8f acc[4] = {};

  for (int k0 = 0; k0 < K; k0 += 32) {
    // ---- A tile (64x32 bf16 = 4KB): async DMA global -> LDS, 16B per lane
    {
      int row = tid >> 2;                 // 0..63
      int kp4 = (tid & 3) << 2;           // k-pair index: 0,4,8,12
      const u32* gp = W32 + (long long)(m0 + row) * Kh + (k0 >> 1) + kp4;
      u32 ldsa = as_base + (u32)((((row << 4) + kp4) << 2));
      asm volatile("global_load_async_to_lds_b128 %0, %1, off"
                   :: "v"(ldsa), "v"(gp) : "memory");
    }
    // ---- B tile (32x128 bf16 = 8KB): wide loads + register k-pair transpose
    {
      int kp = tid >> 4;                  // 0..15 -> rows k=2kp, 2kp+1
      int nb = (tid & 15) << 3;           // 0,8,...,120
      const u16* r0p = X + (long long)(k0 + 2 * kp) * ncols + n0 + nb;
      uint4 lo = *(const uint4*)r0p;              // k row,   n..n+7
      uint4 hi = *(const uint4*)(r0p + ncols);    // k+1 row, n..n+7
      u32 la[4] = {lo.x, lo.y, lo.z, lo.w};
      u32 ha[4] = {hi.x, hi.y, hi.z, hi.w};
#pragma unroll
      for (int j = 0; j < 4; ++j) {
        u32 p0 = (la[j] & 0xFFFFu) | (ha[j] << 16);           // col nb+2j
        u32 p1 = (la[j] >> 16) | (ha[j] & 0xFFFF0000u);       // col nb+2j+1
        Bs[(nb + 2 * j) * 16 + kp]     = p0;
        Bs[(nb + 2 * j + 1) * 16 + kp] = p1;
      }
    }
    // prefetch next K tile of X
    if (k0 + 32 < K && (tid & 3) == 0) {
      int q = tid >> 2;                   // 0..63
      int kq = q >> 1, hh = (q & 1) * 64;
      __builtin_prefetch((const void*)(X + (long long)(k0 + 32 + kq) * ncols + n0 + hh), 0, 1);
    }
    asm volatile("s_wait_asynccnt 0x0" ::: "memory");
    __syncthreads();

    union { v16bf v; u32 w[8]; } fa;
#pragma unroll
    for (int v = 0; v < 8; ++v) {         // A: lanes 0-15 K=0..15, lanes 16-31 K=16..31
      int kk = (v << 1) + (hi16 << 4);
      fa.w[v] = As[(wm * 16 + l15) * 16 + (kk >> 1)];
    }
#pragma unroll
    for (int tn = 0; tn < 4; ++tn) {      // B: lanes 0-15 K=0..15, lanes 16-31 K=16..31
      union { v16bf v; u32 w[8]; } fb;
      int nb = (wn * 64 + tn * 16 + l15) * 16 + (hi16 << 3);
#pragma unroll
      for (int v = 0; v < 8; ++v) fb.w[v] = Bs[nb + v];
      acc[tn] = __builtin_amdgcn_wmma_f32_16x16x32_bf16(false, fa.v, false, fb.v, (short)0, acc[tn], false, false);
    }
    __syncthreads();
  }

#pragma unroll
  for (int v = 0; v < 8; ++v) {           // C/D: row = v + 8*hi16, col = lane&15
    int row = m0 + wm * 16 + v + (hi16 << 3);
    if (row < M) {
      float bv = bias[row];
#pragma unroll
      for (int tn = 0; tn < 4; ++tn) {
        long long c0 = n0 + wn * 64 + tn * 16 + l15;
        float r = acc[tn][v] + bv;
        if (relu) r = fmaxf(r, 0.0f);
        Y[(long long)row * ncols + c0] = f2bf(r);
      }
    }
  }
}

// ---------------------------------------------------------------- post ops

__global__ void k_maxpool(const u16* __restrict__ Y, float* __restrict__ fea,
                          int C, int S, int ncols, int total) {
  int t = blockIdx.x * blockDim.x + threadIdx.x;
  if (t >= total) return;               // fea (B,C,S); Y (C, B*S*ns)
  int b = t / (C * S); int rm = t % (C * S);
  int c = rm / S;      int s  = rm % S;
  const u16* base = Y + (size_t)c * ncols + ((size_t)b * S + s) * PN_NS;
  float m = -1e30f;
#pragma unroll
  for (int j = 0; j < PN_NS; ++j) m = fmaxf(m, bf2f(base[j]));
  fea[t] = m;
}

__global__ void k_unpack(const u16* __restrict__ Y, float* __restrict__ fea,
                         int C, int n, int ncols, int total) {
  int t = blockIdx.x * blockDim.x + threadIdx.x;
  if (t >= total) return;               // fea (B,C,n); Y (C, B*n)
  int b = t / (C * n); int rm = t % (C * n);
  int c = rm / n;      int i  = rm % n;
  fea[t] = bf2f(Y[(size_t)c * ncols + (size_t)b * n + i]);
}

// point_features (B,N,128) = transpose of FP1 output (128, B*N); LDS-tiled for coalescing
__global__ __launch_bounds__(256)
void k_pf_transpose(const u16* __restrict__ Xfp1, float* __restrict__ out, int ncols) {
  __shared__ float T[32 * 128];
  const long long col0 = (long long)blockIdx.x * 32;
  for (int e = threadIdx.x; e < 4096; e += 256) {
    int row = e >> 5, c = e & 31;
    T[c * 128 + row] = bf2f(Xfp1[(long long)row * ncols + col0 + c]);
  }
  __syncthreads();
  for (int e = threadIdx.x; e < 4096; e += 256) {
    int c = e >> 7, row = e & 127;
    out[(col0 + c) * 128 + row] = T[c * 128 + row];
  }
}

// sigmoid cls + scaled offsets from head-GEMM output Yh (4 x ncols) bf16
__global__ void k_finalize(const u16* __restrict__ Yh, float* __restrict__ out, int ncols) {
  int t = blockIdx.x * blockDim.x + threadIdx.x;
  if (t >= ncols) return;
  const size_t O1 = (size_t)PN_B * PN_N * 128;
  const size_t O2 = O1 + (size_t)PN_B * PN_N;
  float a0 = bf2f(Yh[(size_t)0 * ncols + t]);
  float a1 = bf2f(Yh[(size_t)1 * ncols + t]);
  float a2 = bf2f(Yh[(size_t)2 * ncols + t]);
  float ac = bf2f(Yh[(size_t)3 * ncols + t]);
  out[O1 + t] = 1.0f / (1.0f + __expf(-ac));
  out[O2 + (size_t)t * 3 + 0] = a0 * 0.12f;
  out[O2 + (size_t)t * 3 + 1] = a1 * 0.12f;
  out[O2 + (size_t)t * 3 + 2] = a2 * 0.12f;
}

// ---------------------------------------------------------------- host orchestration

struct LayerDesc { int wi; int cout; int cin; };

extern "C" void kernel_launch(void* const* d_in, const int* in_sizes, int n_in,
                              void* d_out, int out_size, void* d_ws, size_t ws_size,
                              hipStream_t stream) {
  (void)in_sizes; (void)n_in; (void)out_size;
  const int B = PN_B, N = PN_N;
  const float* points = (const float*)d_in[0];

  const LayerDesc saL[4][3] = {
    {{1, 32, 6},    {3, 32, 32},   {5, 64, 32}},
    {{7, 64, 67},   {9, 64, 64},   {11, 128, 64}},
    {{13, 128, 131},{15, 128, 128},{17, 256, 128}},
    {{19, 256, 259},{21, 256, 256},{23, 512, 256}},
  };
  const LayerDesc fpL[4][3] = {
    {{25, 256, 768},{27, 256, 256},{-1, 0, 0}},
    {{29, 256, 384},{31, 256, 256},{-1, 0, 0}},
    {{33, 256, 320},{35, 128, 256},{-1, 0, 0}},
    {{37, 128, 128},{39, 128, 128},{41, 128, 128}},
  };
  const LayerDesc sharedL = {43, 128, 128};

  LayerDesc all[32]; int nall = 0;
  for (int i = 0; i < 4; ++i) for (int l = 0; l < 3; ++l) all[nall++] = saL[i][l];
  for (int i = 0; i < 4; ++i) for (int l = 0; l < 3; ++l) if (fpL[i][l].wi >= 0) all[nall++] = fpL[i][l];
  all[nall++] = sharedL;

  char* wsb = (char*)d_ws; size_t off = 0;
  auto alloc = [&](size_t bytes) -> void* {
    off = (off + 255) & ~(size_t)255;
    void* p = wsb + off; off += bytes; return p;
  };

  u16* Wbf[64] = {};
  for (int a = 0; a < nall; ++a) {
    int Mpad = (all[a].cout + 63) & ~63;
    int Kpad = (all[a].cin + 31) & ~31;
    Wbf[all[a].wi] = (u16*)alloc((size_t)Mpad * Kpad * sizeof(u16));
  }

  const int   Slv[4] = {256, 128, 64, 16};
  const float rad[4] = {0.1f, 0.2f, 0.4f, 0.8f};
  const int   Clv[4] = {64, 128, 256, 512};

  float* l0fea = (float*)alloc((size_t)B * 3 * N * 4);
  float* xyzL[4]; float* feaL[4]; int* fpsI[4]; int* gI[4];
  for (int i = 0; i < 4; ++i) {
    xyzL[i] = (float*)alloc((size_t)B * Slv[i] * 3 * 4);
    feaL[i] = (float*)alloc((size_t)B * Clv[i] * Slv[i] * 4);
    fpsI[i] = (int*)alloc((size_t)B * Slv[i] * 4);
    gI[i]   = (int*)alloc((size_t)B * Slv[i] * PN_NS * 4);
  }
  float* l3f = (float*)alloc((size_t)B * 256 * 64 * 4);
  float* l2f = (float*)alloc((size_t)B * 256 * 128 * 4);
  float* l1f = (float*)alloc((size_t)B * 128 * 256 * 4);
  u16*  Whd = (u16*)alloc((size_t)64 * 128 * sizeof(u16));
  float* bhd = (float*)alloc(64 * 4);
  u16*  Yhd = (u16*)alloc((size_t)4 * B * N * sizeof(u16));

  const size_t BIG = (size_t)128 * B * N * sizeof(u16);   // 67 MB
  u16* XBUF = (u16*)alloc(BIG);
  u16* PING = (u16*)alloc(BIG);
  u16* PONG = (u16*)alloc(BIG);
  if (off > ws_size) return;

  // ---- weight prep
  for (int a = 0; a < nall; ++a) {
    int Mpad = (all[a].cout + 63) & ~63;
    int Kpad = (all[a].cin + 31) & ~31;
    int total = Mpad * Kpad;
    k_convert_w<<<(total + 255) / 256, 256, 0, stream>>>(
        (const float*)d_in[all[a].wi], Wbf[all[a].wi], all[a].cout, all[a].cin, Kpad, total);
  }
  k_pack_heads<<<(64 * 128 + 255) / 256, 256, 0, stream>>>(
      (const float*)d_in[45], (const float*)d_in[46],
      (const float*)d_in[47], (const float*)d_in[48], Whd, bhd);

  {
    int total = B * 3 * N;
    k_transpose_l0<<<(total + 255) / 256, 256, 0, stream>>>(points, l0fea, total);
  }

  // ---- SA levels
  const float* curXyz = points; const float* curFea = l0fea;
  int curN = N, curC = 3;
  for (int i = 0; i < 4; ++i) {
    const int S = Slv[i]; const float r2 = rad[i] * rad[i];
    size_t shmem = (size_t)curN * 4 + 512 * 8;
    k_fps<<<B, 512, shmem, stream>>>(curXyz, fpsI[i], curN, S);
    k_gather_xyz<<<(B * S + 255) / 256, 256, 0, stream>>>(curXyz, fpsI[i], xyzL[i], S, curN, B * S);
    k_ballquery<<<(B * S + 255) / 256, 256, 0, stream>>>(curXyz, xyzL[i], gI[i], curN, S, r2, B * S);

    const int ncols = B * S * PN_NS;
    const int Kpad  = (3 + curC + 31) & ~31;
    k_sa_build<<<(ncols + 255) / 256, 256, 0, stream>>>(
        curXyz, xyzL[i], gI[i], curFea, XBUF, curN, S, curC, Kpad, ncols);

    const u16* xin = XBUF; u16* yout = PING;
    for (int l = 0; l < 3; ++l) {
      LayerDesc d = saL[i][l];
      int K = (l == 0) ? Kpad : saL[i][l - 1].cout;
      int Mpad = (d.cout + 63) & ~63;
      dim3 g(ncols / 128, Mpad / 64);
      k_gemm_bias_relu<<<g, 256, 0, stream>>>(
          Wbf[d.wi], (const float*)d_in[d.wi + 1], xin, yout, d.cout, K, ncols, 1);
      xin = yout; yout = (yout == PING) ? PONG : PING;
    }
    int total = B * Clv[i] * S;
    k_maxpool<<<(total + 255) / 256, 256, 0, stream>>>(xin, feaL[i], Clv[i], S, ncols, total);
    curXyz = xyzL[i]; curFea = feaL[i]; curN = S; curC = Clv[i];
  }

  // ---- FP levels
  const float* fpP1[4] = {xyzL[2], xyzL[1], xyzL[0], points};
  const float* fpP2[4] = {xyzL[3], xyzL[2], xyzL[1], xyzL[0]};
  const float* fpF1[4] = {feaL[2], feaL[1], feaL[0], nullptr};
  const float* fpF2[4] = {feaL[3], l3f, l2f, l1f};
  const int fpN[4]  = {64, 128, 256, N};
  const int fpM[4]  = {16, 64, 128, 256};
  const int fpC1[4] = {256, 128, 64, 0};
  const int fpC2[4] = {512, 256, 256, 128};
  float* fpOut[4] = {l3f, l2f, l1f, nullptr};
  const int fpNL[4] = {2, 2, 2, 3};

  const u16* fp1_final = nullptr;
  for (int j = 0; j < 4; ++j) {
    const int n = fpN[j], m = fpM[j], C1 = fpC1[j], C2 = fpC2[j];
    const int ncols = B * n;
    k_fp_build<<<(ncols + 255) / 256, 256, 0, stream>>>(
        fpP1[j], fpP2[j], fpF1[j], fpF2[j], XBUF, n, m, C1, C2, ncols);

    const u16* xin = XBUF; u16* yout = PING;
    int lastC = 0;
    for (int l = 0; l < fpNL[j]; ++l) {
      LayerDesc d = fpL[j][l];
      int K = (l == 0) ? (C1 + C2) : fpL[j][l - 1].cout;
      int Mpad = (d.cout + 63) & ~63;
      dim3 g(ncols / 128, Mpad / 64);
      k_gemm_bias_relu<<<g, 256, 0, stream>>>(
          Wbf[d.wi], (const float*)d_in[d.wi + 1], xin, yout, d.cout, K, ncols, 1);
      xin = yout; yout = (yout == PING) ? PONG : PING;
      lastC = d.cout;
    }
    if (fpOut[j]) {
      int total = B * lastC * n;
      k_unpack<<<(total + 255) / 256, 256, 0, stream>>>(xin, fpOut[j], lastC, n, ncols, total);
    } else {
      fp1_final = xin;   // bf16 (128, B*N), ends in PING
    }
  }

  // ---- shared MLP, heads GEMM, outputs
  {
    const int ncols = B * N;
    u16* shout = (fp1_final == PING) ? PONG : PING;
    dim3 g(ncols / 128, 2);
    k_gemm_bias_relu<<<g, 256, 0, stream>>>(
        Wbf[sharedL.wi], (const float*)d_in[sharedL.wi + 1], fp1_final, shout, 128, 128, ncols, 1);

    dim3 gh(ncols / 128, 1);   // heads: M=4 (3 offsets + 1 cls), no relu
    k_gemm_bias_relu<<<gh, 256, 0, stream>>>(Whd, bhd, shout, Yhd, 4, 128, ncols, 0);

    k_pf_transpose<<<ncols / 32, 256, 0, stream>>>(fp1_final, (float*)d_out, ncols);
    k_finalize<<<(ncols + 255) / 256, 256, 0, stream>>>(Yhd, (float*)d_out, ncols);
  }
}
